// SwinBlock_81028853006538
// MI455X (gfx1250) — matrix operand order repaired and checked by reading, not compiled, not run on hardware
//
#include <hip/hip_runtime.h>
#include <math.h>

// ---------------------------------------------------------------------------
// Types for CDNA5 WMMA / TDM
// ---------------------------------------------------------------------------
typedef __attribute__((ext_vector_type(16))) __bf16       v16bf;
typedef __attribute__((ext_vector_type(8)))  float        v8f;
typedef __attribute__((ext_vector_type(4)))  unsigned int u32x4;
typedef __attribute__((ext_vector_type(4)))  unsigned int tdm_g0_t;
typedef __attribute__((ext_vector_type(8)))  int          tdm_g1_t;
typedef __attribute__((ext_vector_type(4)))  int          tdm_g23_t;
typedef __attribute__((ext_vector_type(8)))  int          tdm_g4_t;

union Frag16 { u32x4 q[2]; v16bf v; };

__device__ __forceinline__ v8f vzero8() {
    v8f z;
    for (int i = 0; i < 8; ++i) z[i] = 0.0f;
    return z;
}

__device__ __forceinline__ v8f wmma_bf16(const v16bf& a, const v16bf& b, const v8f& c) {
    // D = A(16x32 bf16) x B(32x16 bf16) + C(16x16 f32)
    return __builtin_amdgcn_wmma_f32_16x16x32_bf16(
        /*neg_a=*/false, a, /*neg_b=*/false, b,
        /*c_mod=*/(short)0, c, /*reuse_a=*/false, /*reuse_b=*/false);
}

// f32 -> bf16 (round to nearest even), bf16 -> f32, as raw ushort bits
__device__ __forceinline__ unsigned short f2bf(float f) {
    unsigned int u = __float_as_uint(f);
    unsigned int r = u + 0x7FFFu + ((u >> 16) & 1u);
    return (unsigned short)(r >> 16);
}
__device__ __forceinline__ float bf2f(unsigned short h) {
    return __uint_as_float(((unsigned int)h) << 16);
}

// ---------------------------------------------------------------------------
// Tensor Data Mover: 2D tile load (bf16 elements) Global -> LDS.
// Descriptor layout per CDNA5 ISA section 8.3/8.4 (D# groups 0 and 1).
// Issued once per block by wave 0 (TDM ignores EXEC; tracked on TENSORcnt).
// Toolchain uses the 6-argument builtin form:
//   (uint32x4 g0, int32x8 g1, int32x4 g2, int32x4 g3, int32x8 extra, i32 cpol)
// ---------------------------------------------------------------------------
__device__ __forceinline__ void tdm_load_tile_2d_bf16(
    unsigned int lds_byte_addr,       // LDS destination (bytes)
    const void* gtile,                // global byte address of tile start
    unsigned int tensor_d0,           // tensor width  (elements)
    unsigned int tensor_d1,           // tensor height (rows)
    unsigned int tile_d0,             // tile width  (elements)
    unsigned int tile_d1,             // tile height (rows)
    unsigned int stride_d0)           // row stride (elements)
{
    unsigned long long ga = (unsigned long long)gtile;
    tdm_g0_t g0;
    g0[0] = 1u;                                            // count=1, user mode
    g0[1] = lds_byte_addr;                                 // lds_addr
    g0[2] = (unsigned int)(ga & 0xFFFFFFFFu);              // global_addr[31:0]
    g0[3] = (unsigned int)((ga >> 32) & 0x1FFFFFFu)        // global_addr[56:32]
          | (2u << 30);                                    // type=2 ("image")
    tdm_g1_t g1;
    g1[0] = (int)(1u << 16);                               // data_size=1 (2 bytes)
    g1[1] = (int)((tensor_d0 & 0xFFFFu) << 16);            // tensor_dim0[15:0]
    g1[2] = (int)((tensor_d0 >> 16) |
                  ((tensor_d1 & 0xFFFFu) << 16));          // dim0[31:16], dim1[15:0]
    g1[3] = (int)((tensor_d1 >> 16) | (tile_d0 << 16));    // dim1[31:16], tile_dim0
    g1[4] = (int)(tile_d1 & 0xFFFFu);                      // tile_dim1, tile_dim2=0
    g1[5] = (int)stride_d0;                                // dim0_stride[31:0]
    g1[6] = 0;                                             // dim0_stride[47:32], dim1_stride[15:0]
    g1[7] = 0;                                             // dim1_stride[47:16]
    tdm_g23_t gz4 = {0, 0, 0, 0};                          // groups 2/3 unused (2D)
    tdm_g4_t  gz8 = {0, 0, 0, 0, 0, 0, 0, 0};
    __builtin_amdgcn_tensor_load_to_lds(g0, g1, gz4, gz4, gz8, 0);
}

// ---------------------------------------------------------------------------
// Problem constants
// ---------------------------------------------------------------------------
#define DIMC   512
#define HEADS  16
#define HD     32
#define WS_    8
#define SS_    4
#define LWIN   64          // tokens per window
#define NWIN   1024        // 32 x 32 windows
#define NTOK   65536       // NWIN * LWIN
#define IMGHW  256
#define QSCALE 0.17677669529663689f   // 1/sqrt(32)

// shifted-grid region id for the Swin attention mask
__device__ __forceinline__ int regid(int z) {
    return (z < IMGHW - WS_) ? 0 : ((z < IMGHW - SS_) ? 1 : 2);
}

// token (window w, slot t) -> original-image coords (after roll -SS on input)
__device__ __forceinline__ void tok2img(int w, int t, int& oh, int& ow) {
    int gh = (w >> 5) * WS_ + (t >> 3);
    int gw = (w & 31) * WS_ + (t & 7);
    oh = (gh + SS_) & (IMGHW - 1);
    ow = (gw + SS_) & (IMGHW - 1);
}

// ---------------------------------------------------------------------------
// f32 -> bf16 weight conversion
// ---------------------------------------------------------------------------
__global__ __launch_bounds__(256)
void convert_bf16_kernel(const float* __restrict__ src,
                         unsigned short* __restrict__ dst, int n) {
    for (int i = blockIdx.x * 256 + threadIdx.x; i < n; i += gridDim.x * 256)
        dst[i] = f2bf(src[i]);
}

// ---------------------------------------------------------------------------
// LayerNorm over C=512.  GATHER: input is x (C,H,W) via window-partition+roll.
// Otherwise input is o1 (NTOK,512) row-major.  Output bf16 (NTOK,512).
// ---------------------------------------------------------------------------
template<bool GATHER>
__global__ __launch_bounds__(256)
void ln_kernel(const float* __restrict__ src,
               const float* __restrict__ lnw,
               const float* __restrict__ lnb,
               unsigned short* __restrict__ dst) {
    int token = blockIdx.x;
    int tid   = threadIdx.x;
    float v0, v1;
    if (GATHER) {
        int oh, ow;
        tok2img(token >> 6, token & 63, oh, ow);
        size_t p = (size_t)oh * IMGHW + ow;
        v0 = src[(size_t)tid * (IMGHW * IMGHW) + p];
        v1 = src[(size_t)(tid + 256) * (IMGHW * IMGHW) + p];
    } else {
        v0 = src[(size_t)token * DIMC + tid];
        v1 = src[(size_t)token * DIMC + tid + 256];
    }
    float s = v0 + v1, s2 = v0 * v0 + v1 * v1;
    for (int off = 16; off > 0; off >>= 1) {
        s  += __shfl_xor(s,  off, 32);
        s2 += __shfl_xor(s2, off, 32);
    }
    __shared__ float rs[8], rq[8];
    int wave = tid >> 5, lane = tid & 31;
    if (lane == 0) { rs[wave] = s; rq[wave] = s2; }
    __syncthreads();
    if (tid == 0) {
        float a = 0.f, b = 0.f;
        for (int i = 0; i < 8; ++i) { a += rs[i]; b += rq[i]; }
        rs[0] = a; rq[0] = b;
    }
    __syncthreads();
    float mu  = rs[0] * (1.0f / DIMC);
    float var = rq[0] * (1.0f / DIMC) - mu * mu;
    float inv = rsqrtf(var + 1e-5f);
    dst[(size_t)token * DIMC + tid]       = f2bf((v0 - mu) * inv * lnw[tid]       + lnb[tid]);
    dst[(size_t)token * DIMC + tid + 256] = f2bf((v1 - mu) * inv * lnw[tid + 256] + lnb[tid + 256]);
}

// ---------------------------------------------------------------------------
// Tiled bf16 WMMA GEMM:  out = epilogue( A(NTOK x K) * W(K x M) + bias )
// Block tile 128x128, BK=32, 256 threads = 8 waves; wave -> 4x2 16x16 tiles.
// A tile staged by the Tensor Data Mover (TDM); W tile staged transposed by
// the threads (TDM cannot transpose).  Next W tile prefetched into GL2.
// EPI: 0=QKV(bias, scale q)  1=PROJ(bias, gamma1, +x gather -> o1 f32)
//      2=FC1(bias, gelu -> bf16)  3=FC2(bias, gamma2, +o1, scatter to d_out)
// ---------------------------------------------------------------------------
template<int EPI>
__global__ __launch_bounds__(256)
void gemm_bf16_kernel(const unsigned short* __restrict__ A,
                      const unsigned short* __restrict__ Wm,
                      const float* __restrict__ bias,
                      int K, int M,
                      const float* __restrict__ gamma,
                      const float* __restrict__ resid,
                      float* __restrict__ out_f32,
                      unsigned short* __restrict__ out_bf16) {
    __shared__ alignas(16) unsigned short As[128 * 32];   // [row][k]  (TDM dest)
    __shared__ alignas(16) unsigned short Wt[128 * 32];   // [col][k]  (transposed)

    const int tid  = threadIdx.x;
    const int lane = tid & 31;
    const int wave = tid >> 5;
    const int wm   = wave >> 2;         // 0..1 : 64-row slice
    const int wn   = wave & 3;          // 0..3 : 32-col slice
    const int m16  = lane & 15;
    const int hg   = lane >> 4;
    const size_t row0 = (size_t)blockIdx.y * 128;
    const int    n0   = blockIdx.x * 128;

    v8f acc[4][2];
    for (int i = 0; i < 4; ++i)
        for (int j = 0; j < 2; ++j) acc[i][j] = vzero8();

    for (int kk = 0; kk < K; kk += 32) {
        // stage A tile (128x32) via Tensor Data Mover, wave 0 only
        if (tid < 32) {
            if (lane == 0) {
                tdm_load_tile_2d_bf16(
                    (unsigned int)(unsigned long long)&As[0],
                    &A[row0 * K + kk],
                    (unsigned int)K, NTOK,      // tensor dims
                    32u, 128u,                  // tile dims (elems x rows)
                    (unsigned int)K);           // row stride (elems)
            }
            __builtin_amdgcn_s_wait_tensorcnt(0);
        }
        // stage W tile (32x128) transposed into Wt[col][k]
        {
            int k = tid >> 3, part = tid & 7;
            union { u32x4 q; unsigned short s[16]; } wv;
            wv.q = *(const u32x4*)&Wm[(size_t)(kk + k) * M + n0 + part * 16];
            for (int i = 0; i < 16; ++i)
                Wt[(part * 16 + i) * 32 + k] = wv.s[i];
            if (kk + 32 < K)   // GL2 prefetch of next W tile rows
                __builtin_prefetch(&Wm[(size_t)(kk + 32 + k) * M + n0 + part * 16], 0, 1);
        }
        __syncthreads();

        Frag16 fa[4], fb[2];
        for (int ti = 0; ti < 4; ++ti) {
            int ar = wm * 64 + ti * 16 + m16;
            fa[ti].q[0] = *(const u32x4*)&As[ar * 32 + 8 * hg];
            fa[ti].q[1] = *(const u32x4*)&As[ar * 32 + 16 + 8 * hg];
        }
        for (int tj = 0; tj < 2; ++tj) {
            int bc = wn * 32 + tj * 16 + m16;
            fb[tj].q[0] = *(const u32x4*)&Wt[bc * 32 + 8 * hg];
            fb[tj].q[1] = *(const u32x4*)&Wt[bc * 32 + 16 + 8 * hg];
        }
        for (int ti = 0; ti < 4; ++ti)
            for (int tj = 0; tj < 2; ++tj)
                acc[ti][tj] = wmma_bf16(fa[ti].v, fb[tj].v, acc[ti][tj]);
        __syncthreads();
    }

    // epilogue
    for (int ti = 0; ti < 4; ++ti)
        for (int tj = 0; tj < 2; ++tj)
            for (int r = 0; r < 8; ++r) {
                size_t row = row0 + wm * 64 + ti * 16 + r + 8 * hg;
                int    col = n0 + wn * 32 + tj * 16 + m16;
                float  val = acc[ti][tj][r] + bias[col];
                if (EPI == 0) {                       // QKV, scale q
                    if (col < DIMC) val *= QSCALE;
                    out_bf16[row * (size_t)M + col] = f2bf(val);
                } else if (EPI == 1) {                // proj -> o1 (f32)
                    int oh, ow;
                    tok2img((int)(row >> 6), (int)(row & 63), oh, ow);
                    float xv = resid[(size_t)col * (IMGHW * IMGHW) + (size_t)oh * IMGHW + ow];
                    out_f32[row * DIMC + col] = xv + gamma[col] * val;
                } else if (EPI == 2) {                // fc1 + exact GELU
                    val = 0.5f * val * (1.0f + erff(val * 0.70710678118654752f));
                    out_bf16[row * (size_t)M + col] = f2bf(val);
                } else {                              // fc2 + residual + scatter
                    int oh, ow;
                    tok2img((int)(row >> 6), (int)(row & 63), oh, ow);
                    float o1v = resid[row * DIMC + col];
                    out_f32[(size_t)col * (IMGHW * IMGHW) + (size_t)oh * IMGHW + ow] =
                        o1v + gamma[col] * val;
                }
            }
}

// ---------------------------------------------------------------------------
// Windowed attention: one wave32 per block, 4 heads per block (grid = 4*NWIN).
// S = Q K^T (WMMA, K=32) + rel-pos bias + shift mask; softmax; O = P V (K=64).
// ---------------------------------------------------------------------------
__global__ __launch_bounds__(32)
void attn_kernel(const unsigned short* __restrict__ qkv,   // (NTOK, 1536) bf16
                 const float* __restrict__ relTable,       // (225, 16)
                 unsigned short* __restrict__ outO) {      // (NTOK, 512) bf16
    __shared__ alignas(16) unsigned short Qs[LWIN * HD];   // [t][d]
    __shared__ alignas(16) unsigned short Ks[LWIN * HD];   // [t][d]  (== B^T layout)
    __shared__ alignas(16) unsigned short Vt[HD * LWIN];   // [d][t]  (V transposed)
    __shared__ alignas(16) float          Sf[LWIN * LWIN];
    __shared__ alignas(16) unsigned short Ps[LWIN * LWIN];

    const int blk   = blockIdx.x;
    const int w     = blk >> 2;
    const int hbase = (blk & 3) * 4;
    const int lane  = threadIdx.x;
    const int m16   = lane & 15;
    const int hg    = lane >> 4;
    const int wh    = (w >> 5) * WS_;
    const int ww    = (w & 31) * WS_;

    for (int hh = 0; hh < 4; ++hh) {
        const int h = hbase + hh;
        // ---- stage Q, K, V^T in LDS ----
        for (int idx = lane; idx < LWIN * HD; idx += 32) {
            int t = idx >> 5, d = idx & 31;
            const unsigned short* rp =
                qkv + (size_t)(w * LWIN + t) * (3 * DIMC) + h * HD + d;
            Qs[t * HD + d] = rp[0];
            Ks[t * HD + d] = rp[DIMC];
            Vt[d * LWIN + t] = rp[2 * DIMC];
        }
        asm volatile("s_wait_dscnt 0" ::: "memory");

        // ---- S = Q K^T  (4x4 tiles of 16x16, single WMMA each) ----
        Frag16 fa, fb;
        for (int ti = 0; ti < 4; ++ti) {
            int ar = ti * 16 + m16;
            fa.q[0] = *(const u32x4*)&Qs[ar * HD + 8 * hg];
            fa.q[1] = *(const u32x4*)&Qs[ar * HD + 16 + 8 * hg];
            v8f sacc[4];
            for (int tj = 0; tj < 4; ++tj) {
                int bc = tj * 16 + m16;
                fb.q[0] = *(const u32x4*)&Ks[bc * HD + 8 * hg];
                fb.q[1] = *(const u32x4*)&Ks[bc * HD + 16 + 8 * hg];
                sacc[tj] = wmma_bf16(fa.v, fb.v, vzero8());
            }
            // bias + mask, spill to LDS
            for (int tj = 0; tj < 4; ++tj)
                for (int r = 0; r < 8; ++r) {
                    int a = ti * 16 + r + 8 * hg;    // query token
                    int b = tj * 16 + m16;           // key token
                    int ai = a >> 3, aj = a & 7, bi = b >> 3, bj = b & 7;
                    int idx = (ai - bi + 7) * 15 + (aj - bj + 7);
                    float v = sacc[tj][r] + relTable[idx * HEADS + h];
                    int ra = regid(wh + ai) * 3 + regid(ww + aj);
                    int rb = regid(wh + bi) * 3 + regid(ww + bj);
                    Sf[a * LWIN + b] = (ra != rb) ? -1e30f : v;
                }
        }
        asm volatile("s_wait_dscnt 0" ::: "memory");

        // ---- softmax rows, write P (bf16) ----
        for (int row = lane; row < LWIN; row += 32) {
            float mx = -1e30f;
            for (int c = 0; c < LWIN; ++c) mx = fmaxf(mx, Sf[row * LWIN + c]);
            float s = 0.f;
            for (int c = 0; c < LWIN; ++c) {
                float e = __expf(Sf[row * LWIN + c] - mx);
                Sf[row * LWIN + c] = e;
                s += e;
            }
            float inv = 1.0f / s;
            for (int c = 0; c < LWIN; ++c)
                Ps[row * LWIN + c] = f2bf(Sf[row * LWIN + c] * inv);
        }
        asm volatile("s_wait_dscnt 0" ::: "memory");

        // ---- O = P V  (4x2 tiles, K=64 -> two chained WMMAs) ----
        for (int ti = 0; ti < 4; ++ti) {
            v8f oacc[2] = { vzero8(), vzero8() };
            for (int ks = 0; ks < 2; ++ks) {
                int ar = ti * 16 + m16;
                fa.q[0] = *(const u32x4*)&Ps[ar * LWIN + ks * 32 + 8 * hg];
                fa.q[1] = *(const u32x4*)&Ps[ar * LWIN + ks * 32 + 16 + 8 * hg];
                for (int tj = 0; tj < 2; ++tj) {
                    int bc = tj * 16 + m16;
                    fb.q[0] = *(const u32x4*)&Vt[bc * LWIN + ks * 32 + 8 * hg];
                    fb.q[1] = *(const u32x4*)&Vt[bc * LWIN + ks * 32 + 16 + 8 * hg];
                    oacc[tj] = wmma_bf16(fa.v, fb.v, oacc[tj]);
                }
            }
            for (int tj = 0; tj < 2; ++tj)
                for (int r = 0; r < 8; ++r) {
                    int row = ti * 16 + r + 8 * hg;
                    int col = tj * 16 + m16;
                    outO[(size_t)(w * LWIN + row) * DIMC + h * HD + col] =
                        f2bf(oacc[tj][r]);
                }
        }
    }
}

// ---------------------------------------------------------------------------
// Host-side launch
// ---------------------------------------------------------------------------
extern "C" void kernel_launch(void* const* d_in, const int* in_sizes, int n_in,
                              void* d_out, int out_size, void* d_ws, size_t ws_size,
                              hipStream_t stream) {
    (void)in_sizes; (void)n_in; (void)out_size; (void)ws_size;

    const float* x        = (const float*)d_in[0];
    const float* n1w      = (const float*)d_in[1];
    const float* n1b      = (const float*)d_in[2];
    const float* qkv_w    = (const float*)d_in[3];
    const float* qkv_b    = (const float*)d_in[4];
    const float* proj_w   = (const float*)d_in[5];
    const float* proj_b   = (const float*)d_in[6];
    const float* rel_tab  = (const float*)d_in[7];
    const float* gamma1   = (const float*)d_in[8];
    const float* n2w      = (const float*)d_in[9];
    const float* n2b      = (const float*)d_in[10];
    const float* fc1_w    = (const float*)d_in[11];
    const float* fc1_b    = (const float*)d_in[12];
    const float* fc2_w    = (const float*)d_in[13];
    const float* fc2_b    = (const float*)d_in[14];
    const float* gamma2   = (const float*)d_in[15];
    float* out            = (float*)d_out;

    // workspace carve-up (bf16 buffers are ushort)
    char* p = (char*)d_ws;
    unsigned short* xn   = (unsigned short*)p; p += (size_t)NTOK * DIMC * 2;        // 64MB
    unsigned short* qkvb = (unsigned short*)p; p += (size_t)NTOK * 3 * DIMC * 2;    // 192MB
    unsigned short* ao   = (unsigned short*)p; p += (size_t)NTOK * DIMC * 2;        // 64MB
    float*          o1   = (float*)p;          p += (size_t)NTOK * DIMC * 4;        // 128MB
    unsigned short* o1n  = (unsigned short*)p; p += (size_t)NTOK * DIMC * 2;        // 64MB
    unsigned short* hbuf = (unsigned short*)p; p += (size_t)NTOK * 4 * DIMC * 2;    // 256MB
    unsigned short* wqB  = (unsigned short*)p; p += (size_t)DIMC * 3 * DIMC * 2;
    unsigned short* wpB  = (unsigned short*)p; p += (size_t)DIMC * DIMC * 2;
    unsigned short* w1B  = (unsigned short*)p; p += (size_t)DIMC * 4 * DIMC * 2;
    unsigned short* w2B  = (unsigned short*)p; p += (size_t)4 * DIMC * DIMC * 2;

    // 1. weights -> bf16
    convert_bf16_kernel<<<2048, 256, 0, stream>>>(qkv_w, wqB, DIMC * 3 * DIMC);
    convert_bf16_kernel<<<1024, 256, 0, stream>>>(proj_w, wpB, DIMC * DIMC);
    convert_bf16_kernel<<<2048, 256, 0, stream>>>(fc1_w, w1B, DIMC * 4 * DIMC);
    convert_bf16_kernel<<<2048, 256, 0, stream>>>(fc2_w, w2B, 4 * DIMC * DIMC);

    // 2. window-partition + LN1 -> xn (bf16)
    ln_kernel<true><<<NTOK, 256, 0, stream>>>(x, n1w, n1b, xn);

    // 3. QKV GEMM (65536x512 @ 512x1536), q scaled
    gemm_bf16_kernel<0><<<dim3(12, 512), 256, 0, stream>>>(
        xn, wqB, qkv_b, DIMC, 3 * DIMC, nullptr, nullptr, nullptr, qkvb);

    // 4. windowed attention (1024 windows x 16 heads)
    attn_kernel<<<4 * NWIN, 32, 0, stream>>>(qkvb, rel_tab, ao);

    // 5. proj GEMM + gamma1 + residual (gather from x) -> o1 (f32)
    gemm_bf16_kernel<1><<<dim3(4, 512), 256, 0, stream>>>(
        ao, wpB, proj_b, DIMC, DIMC, gamma1, x, o1, nullptr);

    // 6. LN2 -> o1n (bf16)
    ln_kernel<false><<<NTOK, 256, 0, stream>>>(o1, n2w, n2b, o1n);

    // 7. fc1 + GELU -> h (bf16)
    gemm_bf16_kernel<2><<<dim3(16, 512), 256, 0, stream>>>(
        o1n, w1B, fc1_b, DIMC, 4 * DIMC, nullptr, nullptr, nullptr, hbuf);

    // 8. fc2 + gamma2 + o1 residual, window-reverse + roll scatter -> out
    gemm_bf16_kernel<3><<<dim3(4, 512), 256, 0, stream>>>(
        hbuf, w2B, fc2_b, 4 * DIMC, DIMC, gamma2, o1, out, nullptr);
}
